// RobustGuaranteed_54760833024614
// MI455X (gfx1250) — matrix operand order repaired and checked by reading, not compiled, not run on hardware
//
#include <hip/hip_runtime.h>
#include <hip/hip_bf16.h>

typedef __attribute__((ext_vector_type(16))) _Float16 v16h;
typedef __attribute__((ext_vector_type(8)))  float    v8f;

#define BB   128          // batch
#define NN   96           // num deltas
#define DD   3072         // C*H*W
#define KC   10           // classes
#define KSTEPS (DD / 32)  // 96 wmma k-steps
#define ROWS (NN * BB)    // 12288
#define TILES (ROWS / 16) // 768

// ---------------- workspace layout (bytes) ----------------
#define WS_WH      0                          // 96*32*16 f16 = 98304 B
#define WS_WL      98304                      // 98304 B
#define WS_LOGITS  196608                     // ROWS*16 f32  = 786432 B
#define WS_L1      983040                     // ROWS f32 = 49152 B
#define WS_L2      1032192
#define WS_DIFF    1081344                    // end = 1130496 B (~1.08 MB)

__device__ __forceinline__ float clip01(float v) {
    return fminf(fmaxf(v, 0.0f), 1.0f);
}

// split f32 -> f16 hi + f16 lo residual (hi+lo ~ f32 accuracy)
__device__ __forceinline__ void split16(float x, _Float16& h, _Float16& l) {
    h = (_Float16)x;
    l = (_Float16)(x - (float)h);
}

// ---------------------------------------------------------------------------
// Kernel 0: repack W (D x 10, zero-pad cols to 16) into per-lane B fragments,
// as hi/lo f16 pairs.  ISA B layout (16-bit 32x16): lane L -> col n = L%16;
// k = 8*(L/16) + (e%8) + 16*(e/8).  Frag index = (kk*32 + lane)*16 + e  -> one contiguous
// 32 B v16h load per lane per k-step in the GEMM.
// ---------------------------------------------------------------------------
__global__ void pack_w(const float* __restrict__ W,
                       _Float16* __restrict__ Wh, _Float16* __restrict__ Wl) {
    int t = blockIdx.x * blockDim.x + threadIdx.x;   // 0 .. 96*32-1
    if (t >= KSTEPS * 32) return;
    int kk   = t >> 5;
    int lane = t & 31;
    int n    = lane & 15;
    int half = lane >> 4;
    v16h fh, fl;
#pragma unroll
    for (int e = 0; e < 16; ++e) {
        int k = kk * 32 + half * 8 + (e & 7) + (e >> 3) * 16;
        float v = (n < KC) ? W[(size_t)k * KC + n] : 0.0f;
        _Float16 h, l;
        split16(v, h, l);
        fh[e] = h;
        fl[e] = l;
    }
    *reinterpret_cast<v16h*>(Wh + (size_t)t * 16) = fh;
    *reinterpret_cast<v16h*>(Wl + (size_t)t * 16) = fl;
}

// ---------------------------------------------------------------------------
// Kernel 1: logits = clip(imgs + deltas) @ W + b.
// One 32-thread workgroup (one wave) per 16-row tile -> 768 single-wave
// workgroups spread across all WGPs; EXEC is all-ones for every WMMA.
// Split-precision: acc += Ah*Bh + Al*Bh + Ah*Bl  (f32 accumulate).
// ---------------------------------------------------------------------------
__global__ void __launch_bounds__(32)
gemm_logits(const float* __restrict__ imgs, const float* __restrict__ deltas,
            const _Float16* __restrict__ Wh, const _Float16* __restrict__ Wl,
            const float* __restrict__ bias, float* __restrict__ logits) {
    const int lane = threadIdx.x;                    // 0..31
    const int tile = blockIdx.x;                     // 0..767

    const int m    = lane & 15;                      // A row within tile
    const int half = lane >> 4;
    const int r    = tile * 16 + m;                  // global row = n*B + b
    const int bidx = r & (BB - 1);                   // r % 128

    const float* __restrict__ drow = deltas + (size_t)r    * DD;
    const float* __restrict__ irow = imgs   + (size_t)bidx * DD;

    v8f acc = {};

#pragma unroll 2
    for (int kk = 0; kk < KSTEPS; ++kk) {
        const int kbase = kk * 32 + half * 8;        // A: k = kbase + {0..7, 16..23}
        // pull the delta stream ~8 iterations (1 KB) ahead
        __builtin_prefetch(drow + kbase + 256, 0, 0);

        v16h ah, al;
#pragma unroll
        for (int h2 = 0; h2 < 2; ++h2) {
            const int kb = kbase + h2 * 16;
            float4 d0 = *reinterpret_cast<const float4*>(drow + kb);
            float4 d1 = *reinterpret_cast<const float4*>(drow + kb + 4);
            float4 i0 = *reinterpret_cast<const float4*>(irow + kb);
            float4 i1 = *reinterpret_cast<const float4*>(irow + kb + 4);
            float xs[8];
            xs[0] = clip01(d0.x + i0.x);
            xs[1] = clip01(d0.y + i0.y);
            xs[2] = clip01(d0.z + i0.z);
            xs[3] = clip01(d0.w + i0.w);
            xs[4] = clip01(d1.x + i1.x);
            xs[5] = clip01(d1.y + i1.y);
            xs[6] = clip01(d1.z + i1.z);
            xs[7] = clip01(d1.w + i1.w);
#pragma unroll
            for (int e = 0; e < 8; ++e) {
                _Float16 h, l;
                split16(xs[e], h, l);
                ah[h2 * 8 + e] = h;
                al[h2 * 8 + e] = l;
            }
        }
        const size_t foff = ((size_t)(kk * 32 + lane)) * 16;
        v16h bh = *reinterpret_cast<const v16h*>(Wh + foff);
        v16h bl = *reinterpret_cast<const v16h*>(Wl + foff);

        acc = __builtin_amdgcn_wmma_f32_16x16x32_f16(
                  false, ah, false, bh, (short)0, acc, false, false);
        acc = __builtin_amdgcn_wmma_f32_16x16x32_f16(
                  false, al, false, bh, (short)0, acc, false, false);
        acc = __builtin_amdgcn_wmma_f32_16x16x32_f16(
                  false, ah, false, bl, (short)0, acc, false, false);
    }

    // C/D layout: lane L -> col n = L%16; acc[v] -> row M = v + 8*(L/16).
    const int n = lane & 15;
    const float bv = (n < KC) ? bias[n] : 0.0f;
#pragma unroll
    for (int v = 0; v < 8; ++v) {
        int rr = tile * 16 + v + 8 * half;
        logits[(size_t)rr * 16 + n] = acc[v] + bv;
    }
}

// ---------------------------------------------------------------------------
// Kernel 2: per-row log-softmax losses. One thread per (n,b) row.
// ---------------------------------------------------------------------------
__global__ void row_losses(const float* __restrict__ logits,
                           const int* __restrict__ labels,
                           float* __restrict__ l1t, float* __restrict__ l2t,
                           float* __restrict__ dif) {
    int r = blockIdx.x * blockDim.x + threadIdx.x;
    if (r >= ROWS) return;
    int n = r / BB;
    int b = r % BB;

    float lg[KC];
#pragma unroll
    for (int k = 0; k < KC; ++k) lg[k] = logits[(size_t)r * 16 + k];

    float mx = lg[0];
#pragma unroll
    for (int k = 1; k < KC; ++k) mx = fmaxf(mx, lg[k]);
    float se = 0.0f;
#pragma unroll
    for (int k = 0; k < KC; ++k) se += expf(lg[k] - mx);
    float lse = logf(se) + mx;

    int lab = labels[b];
    float llab = lg[0];
#pragma unroll
    for (int k = 0; k < KC; ++k) if (k == lab) llab = lg[k];
    float loss1 = lse - llab;                     // -logp[label]

    // top-2 (ties -> lower index first, matches lax.top_k)
    float best = -3.0e38f, second = -3.0e38f;
    int bi = 0, si = 0;
#pragma unroll
    for (int k = 0; k < KC; ++k) {
        float v = lg[k];
        if (v > best)        { second = best; si = bi; best = v; bi = k; }
        else if (v > second) { second = v; si = k; }
    }
    float lsec = lg[0];
#pragma unroll
    for (int k = 0; k < KC; ++k) if (k == si) lsec = lg[k];
    float loss2 = (bi == lab) ? (lse - lsec) : -10000.0f;

    l1t[(size_t)b * NN + n] = loss1;
    l2t[(size_t)b * NN + n] = loss2;
    dif[(size_t)b * NN + n] = loss1 - loss2;
}

// ---------------------------------------------------------------------------
// Kernel 3: per-b stable rank selection (rank == ind) + final reduction.
// robust_loss = mean_b( mean_n l1[b,n] - l2[b, argsort(diff[b,:])[ind]] ).
// ---------------------------------------------------------------------------
__global__ void finalize(const float* __restrict__ l1t, const float* __restrict__ l2t,
                         const float* __restrict__ dif, const int* __restrict__ ind_p,
                         float* __restrict__ out) {
    __shared__ float sred[BB];
    int b = threadIdx.x;                 // 0..127
    int ind = *ind_p;

    const float* db = dif + (size_t)b * NN;
    const float* l1 = l1t + (size_t)b * NN;
    const float* l2 = l2t + (size_t)b * NN;

    float suml1 = 0.0f;
    for (int j = 0; j < NN; ++j) suml1 += l1[j];

    float selected = 0.0f;
    for (int i = 0; i < NN; ++i) {
        float di = db[i];
        int rank = 0;
        for (int j = 0; j < NN; ++j) {
            float dj = db[j];
            rank += (dj < di) || (dj == di && j < i);   // stable sort rank
        }
        if (rank == ind) selected = l2[i];
    }

    sred[b] = suml1 * (1.0f / NN) - selected;
    __syncthreads();
    for (int s = 64; s > 0; s >>= 1) {
        if (b < s) sred[b] += sred[b + s];
        __syncthreads();
    }
    if (b == 0) out[0] = sred[0] * (1.0f / BB);
}

extern "C" void kernel_launch(void* const* d_in, const int* in_sizes, int n_in,
                              void* d_out, int out_size, void* d_ws, size_t ws_size,
                              hipStream_t stream) {
    const float* imgs   = (const float*)d_in[0];   // (128,3,32,32)
    const float* deltas = (const float*)d_in[1];   // (96,128,3,32,32)
    const float* W      = (const float*)d_in[2];   // (3072,10)
    const float* bias   = (const float*)d_in[3];   // (10,)
    const int*   labels = (const int*)d_in[4];     // (128,)
    const int*   ind    = (const int*)d_in[5];     // scalar (=91)

    char* ws = (char*)d_ws;
    _Float16* Wh     = (_Float16*)(ws + WS_WH);
    _Float16* Wl     = (_Float16*)(ws + WS_WL);
    float*    logits = (float*)(ws + WS_LOGITS);
    float*    l1t    = (float*)(ws + WS_L1);
    float*    l2t    = (float*)(ws + WS_L2);
    float*    dift   = (float*)(ws + WS_DIFF);

    pack_w<<<(KSTEPS * 32 + 255) / 256, 256, 0, stream>>>(W, Wh, Wl);
    gemm_logits<<<TILES, 32, 0, stream>>>(imgs, deltas, Wh, Wl, bias, logits);
    row_losses<<<(ROWS + 255) / 256, 256, 0, stream>>>(logits, labels, l1t, l2t, dift);
    finalize<<<1, BB, 0, stream>>>(l1t, l2t, dift, ind, (float*)d_out);
}
